// TGATLayer_v4_9955734192803
// MI455X (gfx1250) — matrix operand order repaired and checked by reading, not compiled, not run on hardware
//
#include <hip/hip_runtime.h>
#include <hip/hip_bf16.h>
#include <math.h>

typedef __bf16 bf16;
typedef __attribute__((ext_vector_type(16))) __bf16 v16bf;
typedef __attribute__((ext_vector_type(8)))  __bf16 v8bf;
typedef __attribute__((ext_vector_type(8)))  float  v8f;

// Problem dims (fixed by the reference)
#define TK    20     // neighbor slots per node
#define TKP   32     // padded rows (2 x 16 WMMA tiles)
#define TD    128    // model dim
#define THD   4      // heads
#define TDK   32     // per-head dim
#define THID  512    // FFN hidden
#define NTHR  256    // 8 wave32

// ---- dynamic LDS layout (bytes), buffers reused across disjoint phases ----
#define OFF_Z     0        // f32 [32][128]  z (residual, kept live)
#define OFF_A     16384    // f32 [32][128]  generic GEMM output
#define OFF_X     32768    // f32 [32][128]  post-LN x / y
#define OFF_BA    49152    // bf16[32][128]  edge -> q -> attn-out
#define OFF_BB    57344    // bf16[32][128]  k -> x
#define OFF_ZB    65536    // bf16[32][128]  z -> attention probs
#define OFF_VT    73728    // bf16[128][32]  v transposed
#define OFF_BIG   81920    // 32KB: scores f32[4][32][32]  /  h1 bf16[32][512]
#define OFF_SELFQ 114688   // f32 [128]
#define OFF_FVEC  115200   // f32 [256]
#define OFF_OUTV  116224   // f32 [128]
#define OFF_NH    116736   // f32 [20][128] node_h tile (async-prefetched)
#define SMEM_TOTAL 126976

// ---------------- CDNA5 async global->LDS (ASYNCcnt) ----------------
// Raw LDS byte offset of a generic pointer known to alias LDS.
__device__ __forceinline__ unsigned lds_offset_of(const void* p) {
  return (unsigned)(unsigned long long)(__attribute__((address_space(3))) const char*)p;
}

// GV-mode async copy: LDS[lds_off] = MEM[gaddr], 16 bytes per lane.
__device__ __forceinline__ void async_ld_b128(unsigned lds_off, const void* gaddr) {
  asm volatile("global_load_async_to_lds_b128 %0, %1, off"
               :
               : "v"(lds_off), "v"(gaddr)
               : "memory");
}

__device__ __forceinline__ void wait_asynccnt0() {
  asm volatile("s_wait_asynccnt 0x0" ::: "memory");
}

// ---------------- WMMA helpers ----------------
// A fragment, 16x32 bf16 tile from row-major LDS (ldk = row stride, contiguous K)
// ISA 16-bit A layout: lanes0-15 M=0..15 hold K {k0..k0+7, k0+16..k0+23};
// lanes16-31 hold K {k0+8..k0+15, k0+24..k0+31}.
__device__ __forceinline__ v16bf frag_A(const bf16* base, int ldk, int row0, int k0) {
  const int lane = threadIdx.x & 31;
  const bf16* p = base + (row0 + (lane & 15)) * ldk + k0 + ((lane & 16) ? 8 : 0);
  v8bf lo = *(const v8bf*)(p);
  v8bf hi = *(const v8bf*)(p + 16);
  return __builtin_shufflevector(lo, hi, 0,1,2,3,4,5,6,7,8,9,10,11,12,13,14,15);
}

// B fragment, 32x16 bf16 tile where the source is stored N-major (srcT[n][k]):
// lanes0-15 hold K {k0..k0+7, k0+16..k0+23} at N=n0+lane; lanes16-31 hold K {k0+8..k0+15, k0+24..k0+31}.
__device__ __forceinline__ v16bf frag_BT(const bf16* srcT, int ldk, int n0, int k0) {
  const int lane = threadIdx.x & 31;
  const bf16* p = srcT + (n0 + (lane & 15)) * ldk + k0 + ((lane & 16) ? 8 : 0);
  v8bf lo = *(const v8bf*)(p);
  v8bf hi = *(const v8bf*)(p + 16);
  return __builtin_shufflevector(lo, hi, 0,1,2,3,4,5,6,7,8,9,10,11,12,13,14,15);
}

__device__ __forceinline__ v8f wmma_bf16(v16bf a, v16bf b, v8f c) {
  // (neg_a, A, neg_b, B, c_mod, C, reuse_a, reuse_b)
  return __builtin_amdgcn_wmma_f32_16x16x32_bf16(false, a, false, b, (short)0, c, false, false);
}

// Generic 32xNOUT = (32xKTOT)*(KTOTxNOUT) GEMM, f32 output (+optional bias).
template <int KTOT, int NOUT>
__device__ __forceinline__ void gemm_f32(const bf16* __restrict__ A,
                                         const bf16* __restrict__ Wt,
                                         float* __restrict__ outp,
                                         const float* __restrict__ bias) {
  const int wave = threadIdx.x >> 5;
  const int lane = threadIdx.x & 31;
  constexpr int NT = 2 * (NOUT / 16);
  for (int t = wave; t < NT; t += 8) {
    const int mt = t & 1, nt = t >> 1;
    v8f acc = {0.f, 0.f, 0.f, 0.f, 0.f, 0.f, 0.f, 0.f};
#pragma unroll
    for (int k0 = 0; k0 < KTOT; k0 += 32)
      acc = wmma_bf16(frag_A(A, KTOT, mt * 16, k0), frag_BT(Wt, KTOT, nt * 16, k0), acc);
    const int nn = nt * 16 + (lane & 15);
    const int m0 = mt * 16 + ((lane & 16) ? 8 : 0);
    const float bv = bias ? bias[nn] : 0.f;
#pragma unroll
    for (int r = 0; r < 8; ++r) outp[(m0 + r) * NOUT + nn] = acc[r] + bv;
  }
}

// Same, but epilogue = bias + relu, bf16 output (for FFN hidden).
template <int KTOT, int NOUT>
__device__ __forceinline__ void gemm_relu_bf16(const bf16* __restrict__ A,
                                               const bf16* __restrict__ Wt,
                                               bf16* __restrict__ outp,
                                               const float* __restrict__ bias) {
  const int wave = threadIdx.x >> 5;
  const int lane = threadIdx.x & 31;
  constexpr int NT = 2 * (NOUT / 16);
  for (int t = wave; t < NT; t += 8) {
    const int mt = t & 1, nt = t >> 1;
    v8f acc = {0.f, 0.f, 0.f, 0.f, 0.f, 0.f, 0.f, 0.f};
#pragma unroll
    for (int k0 = 0; k0 < KTOT; k0 += 32)
      acc = wmma_bf16(frag_A(A, KTOT, mt * 16, k0), frag_BT(Wt, KTOT, nt * 16, k0), acc);
    const int nn = nt * 16 + (lane & 15);
    const int m0 = mt * 16 + ((lane & 16) ? 8 : 0);
    const float bv = bias[nn];
#pragma unroll
    for (int r = 0; r < 8; ++r)
      outp[(m0 + r) * NOUT + nn] = (bf16)fmaxf(acc[r] + bv, 0.f);
  }
}

__device__ __forceinline__ float gelu_exact(float x) {
  return 0.5f * x * (1.f + erff(x * 0.70710678118654752f));
}

// Per-row LayerNorm over 128 cols; one wave32 per row (lane covers 4 cols).
__device__ __forceinline__ void layer_norm_rows(const float* __restrict__ src,
                                                const float* __restrict__ resid,
                                                const float* __restrict__ g,
                                                const float* __restrict__ be,
                                                float* __restrict__ dst_f,
                                                bf16* __restrict__ dst_b,
                                                int nrows, int ldn) {
  const int wave = threadIdx.x >> 5;
  const int lane = threadIdx.x & 31;
  for (int r = wave; r < nrows; r += 8) {
    float v[4];
    float s = 0.f;
#pragma unroll
    for (int i = 0; i < 4; ++i) {
      const int c = lane + 32 * i;
      float x = src[r * ldn + c] + (resid ? resid[r * ldn + c] : 0.f);
      v[i] = x;
      s += x;
    }
#pragma unroll
    for (int off = 16; off > 0; off >>= 1) s += __shfl_xor(s, off, 32);
    const float mean = s * (1.f / 128.f);
    float vs = 0.f;
#pragma unroll
    for (int i = 0; i < 4; ++i) { float d = v[i] - mean; vs += d * d; }
#pragma unroll
    for (int off = 16; off > 0; off >>= 1) vs += __shfl_xor(vs, off, 32);
    const float inv = rsqrtf(vs * (1.f / 128.f) + 1e-5f);
#pragma unroll
    for (int i = 0; i < 4; ++i) {
      const int c = lane + 32 * i;
      float y = (v[i] - mean) * inv * g[c] + be[c];
      if (dst_f) dst_f[r * ldn + c] = y;
      if (dst_b) dst_b[r * ldn + c] = (bf16)y;
    }
  }
}

// ---------------- weight pre-pass: f32 [Kin][Nout] -> bf16 transposed [Nout][Kin]
__global__ void wt_bf16_transpose(const float* __restrict__ w, bf16* __restrict__ wt,
                                  int kin, int nout) {
  int i = blockIdx.x * blockDim.x + threadIdx.x;
  if (i < kin * nout) {
    int o = i % nout, k = i / nout;
    wt[o * kin + k] = (bf16)w[i];
  }
}

// ---------------- main per-node kernel ----------------
__global__ __launch_bounds__(NTHR)
void tgat_layer_kernel(
    const float* __restrict__ node_h, const float* __restrict__ tt,
    const float* __restrict__ t_now, const float* __restrict__ edge,
    const float* __restrict__ freq, const float* __restrict__ phs,
    const float* __restrict__ edge_b, const float* __restrict__ wqs_src,
    const float* __restrict__ attn_b, const float* __restrict__ attn_g,
    const float* __restrict__ attn_be, const float* __restrict__ ffn_b1,
    const float* __restrict__ ffn_b2, const float* __restrict__ ffn_g,
    const float* __restrict__ ffn_be, const float* __restrict__ fuse_w,
    const float* __restrict__ fuse_b, const float* __restrict__ out_g,
    const float* __restrict__ out_be,
    const bf16* __restrict__ wtEdge, const bf16* __restrict__ wtQ,
    const bf16* __restrict__ wtK, const bf16* __restrict__ wtV,
    const bf16* __restrict__ wtAttn, const bf16* __restrict__ wtF1,
    const bf16* __restrict__ wtF2, float* __restrict__ outp) {
  extern __shared__ __align__(16) char smem[];
  float* s_z  = (float*)(smem + OFF_Z);
  float* s_a  = (float*)(smem + OFF_A);
  float* s_x  = (float*)(smem + OFF_X);
  bf16*  s_bA = (bf16*)(smem + OFF_BA);
  bf16*  s_bB = (bf16*)(smem + OFF_BB);
  bf16*  s_zb = (bf16*)(smem + OFF_ZB);
  bf16*  s_vT = (bf16*)(smem + OFF_VT);
  float* s_sc = (float*)(smem + OFF_BIG);
  bf16*  s_h1 = (bf16*)(smem + OFF_BIG);
  float* s_sq = (float*)(smem + OFF_SELFQ);
  float* s_fv = (float*)(smem + OFF_FVEC);
  float* s_ov = (float*)(smem + OFF_OUTV);
  float* s_nh = (float*)(smem + OFF_NH);

  const int n = blockIdx.x;
  const int tid = threadIdx.x;
  const int wave = tid >> 5;
  const int lane = tid & 31;
  const float tn = t_now[0];

  // A0: kick off async HBM->LDS copy of this node's node_h tile (10 KB).
  //     Overlaps with the edge-feature load + edge_fc GEMM below (ASYNCcnt).
  {
    const unsigned nh_lds = lds_offset_of(s_nh);
    const char* gsrc = (const char*)(node_h + (size_t)n * TK * TD);
    for (int i = tid; i < (TK * TD * 4) / 16; i += NTHR)  // 640 x 16B chunks
      async_ld_b128(nh_lds + i * 16, gsrc + (size_t)i * 16);
  }

  // A: edge features -> bf16 LDS (rows >= 20 zero-padded)
  for (int i = tid; i < TKP * TD; i += NTHR) {
    const int r = i >> 7, c = i & 127;
    s_bA[i] = (r < TK) ? (bf16)edge[(n * TK + r) * TD + c] : (bf16)0.f;
  }
  __syncthreads();

  // B: edge_fc GEMM (+bias)
  gemm_f32<TD, TD>(s_bA, wtEdge, s_a, edge_b);
  wait_asynccnt0();  // node_h tile resident before the barrier releases phase C
  __syncthreads();

  // C: z = node_h + gelu(edge_fc) + cos((t_now - t)*freq + phase)
  for (int i = tid; i < TKP * TD; i += NTHR) {
    const int r = i >> 7, c = i & 127;
    if (r < TK) {
      const float dt = tn - tt[n * TK + r];
      const float zv = s_nh[i] + gelu_exact(s_a[i]) + cosf(dt * freq[c] + phs[c]);
      s_z[i] = zv;
      s_zb[i] = (bf16)zv;
    } else {
      s_zb[i] = (bf16)0.f;
    }
  }
  __syncthreads();

  // D: self-row projection z[19] @ w_qs_src (VALU, M=1) + Q GEMM (WMMA)
  if (tid < TD) {
    float acc = 0.f;
    for (int i2 = 0; i2 < TD; ++i2)
      acc += s_z[(TK - 1) * TD + i2] * wqs_src[i2 * TD + tid];
    s_sq[tid] = acc;
  }
  gemm_f32<TD, TD>(s_zb, wtQ, s_a, nullptr);
  __syncthreads();

  // E: q -> bf16, last slot replaced by self_q (reference uses self_q for q,k,v)
  for (int i = tid; i < TKP * TD; i += NTHR) {
    const int r = i >> 7, c = i & 127;
    const float v = (r < TK - 1) ? s_a[i] : ((r == TK - 1) ? s_sq[c] : 0.f);
    s_bA[i] = (bf16)v;
  }
  __syncthreads();

  // F/G: K GEMM + bf16 convert (row-major, used as transposed B in scores)
  gemm_f32<TD, TD>(s_zb, wtK, s_a, nullptr);
  __syncthreads();
  for (int i = tid; i < TKP * TD; i += NTHR) {
    const int r = i >> 7, c = i & 127;
    const float v = (r < TK - 1) ? s_a[i] : ((r == TK - 1) ? s_sq[c] : 0.f);
    s_bB[i] = (bf16)v;
  }
  __syncthreads();

  // H/I: V GEMM + transposed bf16 store vT[d][k]
  gemm_f32<TD, TD>(s_zb, wtV, s_a, nullptr);
  __syncthreads();
  for (int i = tid; i < TD * TKP; i += NTHR) {
    const int c = i >> 5, r = i & 31;
    const float v = (r < TK - 1) ? s_a[r * TD + c] : ((r == TK - 1) ? s_sq[c] : 0.f);
    s_vT[c * TKP + r] = (bf16)v;
  }
  __syncthreads();

  // J: per-head attention scores Q@K^T (K-dim = 32, one WMMA per tile);
  //    also zero the (reused) prob buffer
  for (int i = tid; i < TKP * TD; i += NTHR) s_zb[i] = (bf16)0.f;
  for (int job = wave; job < 16; job += 8) {
    const int h = job >> 2, mt = (job >> 1) & 1, nt = job & 1;
    v8f acc = {0.f, 0.f, 0.f, 0.f, 0.f, 0.f, 0.f, 0.f};
    acc = wmma_bf16(frag_A(s_bA, TD, mt * 16, h * TDK),
                    frag_BT(s_bB, TD, nt * 16, h * TDK), acc);
    const int nn = nt * 16 + (lane & 15);
    const int m0 = mt * 16 + ((lane & 16) ? 8 : 0);
#pragma unroll
    for (int r2 = 0; r2 < 8; ++r2) s_sc[h * 1024 + (m0 + r2) * TKP + nn] = acc[r2];
  }
  __syncthreads();

  // K: masked softmax over 20 keys, scale 1/sqrt(32); probs -> bf16
  if (tid < THD * TK) {
    const int h = tid / TK, r = tid % TK;
    const float* row = s_sc + h * 1024 + r * TKP;
    const float scale = 0.17677669529663687f;
    float mx = -1e30f;
    for (int j = 0; j < TK; ++j) mx = fmaxf(mx, row[j] * scale);
    float e[TK];
    float sum = 0.f;
    for (int j = 0; j < TK; ++j) { e[j] = expf(row[j] * scale - mx); sum += e[j]; }
    const float inv = 1.f / sum;
    for (int j = 0; j < TK; ++j) s_zb[h * 1024 + r * TKP + j] = (bf16)(e[j] * inv);
  }
  __syncthreads();

  // L: out = P @ V per head (K-dim = 32)
  for (int job = wave; job < 16; job += 8) {
    const int h = job >> 2, mt = (job >> 1) & 1, nt = job & 1;
    v8f acc = {0.f, 0.f, 0.f, 0.f, 0.f, 0.f, 0.f, 0.f};
    acc = wmma_bf16(frag_A(s_zb + h * 1024, TKP, mt * 16, 0),
                    frag_BT(s_vT + h * TDK * TKP, TKP, nt * 16, 0), acc);
    const int nn = h * TDK + nt * 16 + (lane & 15);
    const int m0 = mt * 16 + ((lane & 16) ? 8 : 0);
#pragma unroll
    for (int r2 = 0; r2 < 8; ++r2) s_a[(m0 + r2) * TD + nn] = acc[r2];
  }
  __syncthreads();

  // M: attn-out -> bf16
  for (int i = tid; i < TKP * TD; i += NTHR) s_bA[i] = (bf16)s_a[i];
  __syncthreads();

  // N: attn_fc GEMM (+bias)
  gemm_f32<TD, TD>(s_bA, wtAttn, s_a, attn_b);
  __syncthreads();

  // O: x = LN(attn_out + z); bf16 copy for FFN, pad rows zero
  layer_norm_rows(s_a, s_z, attn_g, attn_be, s_x, s_bB, TK, TD);
  for (int i = TK * TD + tid; i < TKP * TD; i += NTHR) s_bB[i] = (bf16)0.f;
  __syncthreads();

  // P: FFN1 (bias + relu, bf16 hidden)
  gemm_relu_bf16<TD, THID>(s_bB, wtF1, s_h1, ffn_b1);
  __syncthreads();

  // Q: FFN2 (+bias)
  gemm_f32<THID, TD>(s_h1, wtF2, s_a, ffn_b2);
  __syncthreads();

  // R: y = LN(ffn + x) -> s_x
  layer_norm_rows(s_a, s_x, ffn_g, ffn_be, s_x, nullptr, TK, TD);
  __syncthreads();

  // S: fuse vector = [y[last], mean(y[neighbors])]
  if (tid < TD) {
    float m = 0.f;
    for (int r = 0; r < TK - 1; ++r) m += s_x[r * TD + tid];
    s_fv[tid] = s_x[(TK - 1) * TD + tid];
    s_fv[TD + tid] = m * (1.f / (TK - 1));
  }
  __syncthreads();

  // T: fea2node (M=1, VALU) + gelu + residual with node_h_src last slot (from LDS)
  if (tid < TD) {
    float acc = fuse_b[tid];
    for (int i2 = 0; i2 < 2 * TD; ++i2) acc += s_fv[i2] * fuse_w[i2 * TD + tid];
    s_ov[tid] = gelu_exact(acc) + s_nh[(TK - 1) * TD + tid];
  }
  __syncthreads();

  // U: final LN -> global
  layer_norm_rows(s_ov, nullptr, out_g, out_be, outp + n * TD, nullptr, 1, TD);
}

// ---------------- host launch ----------------
extern "C" void kernel_launch(void* const* d_in, const int* in_sizes, int n_in,
                              void* d_out, int out_size, void* d_ws, size_t ws_size,
                              hipStream_t stream) {
  const float* node_h  = (const float*)d_in[0];
  const float* tt      = (const float*)d_in[1];
  const float* t_now   = (const float*)d_in[2];
  const float* edge    = (const float*)d_in[3];
  const float* freq    = (const float*)d_in[4];
  const float* phs     = (const float*)d_in[5];
  const float* edge_w  = (const float*)d_in[6];
  const float* edge_b  = (const float*)d_in[7];
  const float* w_qs    = (const float*)d_in[8];
  const float* w_ks    = (const float*)d_in[9];
  const float* w_vs    = (const float*)d_in[10];
  const float* wqs_src = (const float*)d_in[11];
  const float* attn_w  = (const float*)d_in[12];
  const float* attn_b  = (const float*)d_in[13];
  const float* attn_g  = (const float*)d_in[14];
  const float* attn_be = (const float*)d_in[15];
  const float* ffn_w1  = (const float*)d_in[16];
  const float* ffn_b1  = (const float*)d_in[17];
  const float* ffn_w2  = (const float*)d_in[18];
  const float* ffn_b2  = (const float*)d_in[19];
  const float* ffn_g   = (const float*)d_in[20];
  const float* ffn_be  = (const float*)d_in[21];
  const float* fuse_w  = (const float*)d_in[22];
  const float* fuse_b  = (const float*)d_in[23];
  const float* out_g   = (const float*)d_in[24];
  const float* out_be  = (const float*)d_in[25];

  const int N = in_sizes[0] / (TK * TD);
  bf16* ws = (bf16*)d_ws;
  // bf16-transposed weight table in workspace
  const size_t oEdge = 0, oQ = 16384, oK = 32768, oV = 49152, oAttn = 65536,
               oF1 = 81920, oF2 = 147456;

  auto tp = [&](const float* w, size_t off, int kin, int nout) {
    const int total = kin * nout;
    wt_bf16_transpose<<<(total + 255) / 256, 256, 0, stream>>>(w, ws + off, kin, nout);
  };
  tp(edge_w, oEdge, TD, TD);
  tp(w_qs,  oQ,  TD, TD);
  tp(w_ks,  oK,  TD, TD);
  tp(w_vs,  oV,  TD, TD);
  tp(attn_w, oAttn, TD, TD);
  tp(ffn_w1, oF1, TD, THID);
  tp(ffn_w2, oF2, THID, TD);

  (void)hipFuncSetAttribute(reinterpret_cast<const void*>(tgat_layer_kernel),
                            hipFuncAttributeMaxDynamicSharedMemorySize, SMEM_TOTAL);

  tgat_layer_kernel<<<N, NTHR, SMEM_TOTAL, stream>>>(
      node_h, tt, t_now, edge, freq, phs, edge_b, wqs_src, attn_b, attn_g,
      attn_be, ffn_b1, ffn_b2, ffn_g, ffn_be, fuse_w, fuse_b, out_g, out_be,
      ws + oEdge, ws + oQ, ws + oK, ws + oV, ws + oAttn, ws + oF1, ws + oF2,
      (float*)d_out);
}
